// MCMCDA_51436528337617
// MI455X (gfx1250) — compile-verified, matrix operand order checked by reading, not run on hardware
//
#include <hip/hip_runtime.h>
#include <hip/hip_bf16.h>
#include <hip/hip_fp16.h>

#define ITER_MCMC 100000
#define ITER_DA   10000
#define MAX_ATT   60000
#define NOBS      256
#define TARGET_ACC 0.234f

typedef __attribute__((ext_vector_type(2)))  float v2f;
typedef __attribute__((ext_vector_type(8)))  float v8f;
typedef __attribute__((ext_vector_type(16))) _Float16 v16h;

// ---------------------------------------------------------------------------
// Kernel 1: basis vectors + 10 Gram sufficient statistics (one block, 256 thr)
// ws layout (floats): [0..15] gram scalars, [16..271] s, [272..527] c,
// [528..783] s2
// ---------------------------------------------------------------------------
__global__ void k_basis_gram(const float* __restrict__ obs_loc,
                             const float* __restrict__ obs_val,
                             float* __restrict__ ws) {
  __shared__ float red[10][256];
  const int i = threadIdx.x;
  const float PI = 3.14159265358979323846f;
  float x = obs_loc[i], y = obs_val[i];
  float s  = __sinf(PI * x);
  float c  = __cosf(PI * x);
  float s2 = __sinf(2.0f * PI * x);
  ws[16 + i]        = s;
  ws[16 + 256 + i]  = c;
  ws[16 + 512 + i]  = s2;
  red[0][i] = y * y;   red[1][i] = y * s;   red[2][i] = y * c;
  red[3][i] = y * s2;  red[4][i] = s * s;   red[5][i] = c * c;
  red[6][i] = s2 * s2; red[7][i] = s * c;   red[8][i] = s * s2;
  red[9][i] = c * s2;
  __syncthreads();
  for (int st = 128; st > 0; st >>= 1) {
    if (i < st) {
#pragma unroll
      for (int j = 0; j < 10; ++j) red[j][i] += red[j][i + st];
    }
    __syncthreads();
  }
  if (i < 10) ws[i] = red[i][0];
}

// ---------------------------------------------------------------------------
// Kernel 2: zero acc_list + theta_inner regions of d_out (30000 floats)
// ---------------------------------------------------------------------------
__global__ void k_zero(float* __restrict__ p, int n) {
  int i = blockIdx.x * blockDim.x + threadIdx.x;
  if (i < n) p[i] = 0.0f;
}

// ---------------------------------------------------------------------------
// Kernel 3: the serial Markov chain. One lane. Per-step cost is a handful of
// FMAs on the 10 Gram scalars plus one v_exp_f32 -- the minimal critical path.
// Writes acc_list[mh] and compacted proposals into theta_inner (also used as
// the GEMM A-matrix; unfilled rows stay zero => zero predictions, matching
// the reference's masked-scatter-into-zeros semantics).
// ---------------------------------------------------------------------------
__global__ void k_chain(const float* __restrict__ ws,
                        const float* __restrict__ theta0,
                        const float* __restrict__ eps_o,
                        const float* __restrict__ u_o,
                        const float* __restrict__ eps_da,
                        const float* __restrict__ u_da,
                        float* __restrict__ acc_list,
                        float* __restrict__ theta_inner) {
  if (threadIdx.x != 0 || blockIdx.x != 0) return;
  const float Syy = ws[0], Sys = ws[1], Syc = ws[2], Sy2 = ws[3];
  const float Sss = ws[4], Scc = ws[5], S22 = ws[6];
  const float Ssc = ws[7], Ss2 = ws[8], Sc2 = ws[9];

  // log-posterior, outer (coarse) model: quadratic form in (t0, t1)
  auto lpo = [&](float t0, float t1) -> float {
    float q = Syy - 2.0f * (t0 * Sys + t1 * Syc)
            + t0 * t0 * Sss + 2.0f * t0 * t1 * Ssc + t1 * t1 * Scc;
    return -0.5f * (t0 * t0 + t1 * t1) - 2.0f * q;  // 0.5/0.25 = 2
  };
  // log-posterior, inner (fine) model: quadratic form in (t0, t1, g)
  auto lpi = [&](float t0, float t1) -> float {
    float g = 0.05f * t0 * t1;
    float q = Syy - 2.0f * (t0 * Sys + t1 * Syc + g * Sy2)
            + t0 * t0 * Sss + t1 * t1 * Scc + g * g * S22
            + 2.0f * (t0 * t1 * Ssc + t0 * g * Ss2 + t1 * g * Sc2);
    return -0.5f * (t0 * t0 + t1 * t1) - 2.0f * q;
  };

  float th0 = theta0[0], th1 = theta0[1];
  float dt = 0.1f;
  float lcur = lpo(th0, th1);

  // ---- Stage 1: adaptive RW-MH ----
  for (int i = 0; i < ITER_MCMC; ++i) {
    float p0 = th0 + dt * eps_o[2 * i];
    float p1 = th1 + dt * eps_o[2 * i + 1];
    float lp = lpo(p0, p1);
    float a  = __expf(fminf(lp - lcur, 0.0f));
    if (u_o[i] < a) { th0 = p0; th1 = p1; lcur = lp; }
    dt = dt + dt * (a - TARGET_ACC) / ((float)i + 1.0f);
  }

  // ---- Stage 2: delayed acceptance, compacted writes ----
  int mh = 0;
  for (int t = 0; t < MAX_ATT; ++t) {
    float p0 = th0 + dt * eps_da[2 * t];
    float p1 = th1 + dt * eps_da[2 * t + 1];
    float lo = lcur;
    float lp = lpo(p0, p1);
    float a  = __expf(fminf(lp - lo, 0.0f));
    bool active = (u_da[2 * t] < a) && (mh < ITER_DA);
    float a2 = __expf(fminf(lpi(p0, p1) - lpi(th0, th1) + lo - lp, 0.0f));
    bool inner = active && (u_da[2 * t + 1] < a2);
    if (active) {
      acc_list[mh] = inner ? 1.0f : 0.0f;
      theta_inner[2 * mh]     = p0;
      theta_inner[2 * mh + 1] = p1;
      ++mh;
    }
    if (inner) { th0 = p0; th1 = p1; lcur = lp; }
  }
}

// ---------------------------------------------------------------------------
// Kernel 4: lik_nn / lik_sol as a (10000 x 4) @ (4 x 256) WMMA GEMM.
// One wave per 16x16 output tile pair; K=4 f32 WMMA. A fragment per ISA:
//   lanes 0-15 : v0=A[m][K0]=p0, v1=A[m][K1]=p1
//   lanes 16-31: v0=A[m][K2]=0.05*p0*p1, v1=A[m][K3]=0
// B fragment (4x16, mirrored): lanes 0-15 hold K0/K1 rows, lanes 16-31 K2/K3.
// ---------------------------------------------------------------------------
__global__ __launch_bounds__(256) void k_wmma_preds(
    const float* __restrict__ P,       // theta_inner region: [10000][2]
    const float* __restrict__ basis,   // s[256], c[256], s2[256]
    float* __restrict__ lik_nn,
    float* __restrict__ lik_sol) {
  const int lane  = threadIdx.x & 31;
  const int wave  = blockIdx.x * (blockDim.x >> 5) + (threadIdx.x >> 5);
  const int mtile = wave >> 4;     // 0..624
  const int ntile = wave & 15;     // 0..15
  const int half  = lane & 15;
  const bool hi   = lane >= 16;

  const int m = mtile * 16 + half;
  const float p0 = P[2 * m];
  const float p1 = P[2 * m + 1];

  const int n = ntile * 16 + half;
  const float sb  = basis[n];
  const float cb  = basis[256 + n];
  const float s2b = basis[512 + n];

#if __has_builtin(__builtin_amdgcn_wmma_f32_16x16x4_f32)
  v2f a;
  a.x = hi ? 0.05f * p0 * p1 : p0;   // K2 : K0
  a.y = hi ? 0.0f            : p1;   // K3 : K1
  v2f b_sol, b_nn;
  b_sol.x = hi ? s2b  : sb;          // K2 : K0
  b_sol.y = hi ? 0.0f : cb;          // K3 : K1
  b_nn.x  = hi ? 0.0f : sb;          // outer model has no s2 term
  b_nn.y  = hi ? 0.0f : cb;
  v8f c0 = {}, c1 = {};
  v8f dnn  = __builtin_amdgcn_wmma_f32_16x16x4_f32(
      false, a, false, b_nn,  (short)0, c0, false, false);
  v8f dsol = __builtin_amdgcn_wmma_f32_16x16x4_f32(
      false, a, false, b_sol, (short)0, c1, false, false);
#else
  // Fallback: f16 16x16x32 WMMA (codegen-confirmed builtin). K=0..3 used,
  // hi-lane fragment slots map to K 8..15 and 24..31 (A and B alike) and stay zero.
  v16h ah = {}, bs = {}, bn = {};
  if (!hi) {
    ah[0] = (_Float16)p0; ah[1] = (_Float16)p1;
    ah[2] = (_Float16)(0.05f * p0 * p1);
    bs[0] = (_Float16)sb; bs[1] = (_Float16)cb; bs[2] = (_Float16)s2b;
    bn[0] = (_Float16)sb; bn[1] = (_Float16)cb;
  }
  v8f c0 = {}, c1 = {};
  v8f dnn  = __builtin_amdgcn_wmma_f32_16x16x32_f16(
      false, ah, false, bn, (short)0, c0, false, false);
  v8f dsol = __builtin_amdgcn_wmma_f32_16x16x32_f16(
      false, ah, false, bs, (short)0, c1, false, false);
#endif

  // C/D layout: VGPR r -> (M = r + (hi?8:0), N = half)
  const int col = ntile * 16 + half;
#pragma unroll
  for (int r = 0; r < 8; ++r) {
    const int row = mtile * 16 + r + (hi ? 8 : 0);
    lik_nn [row * NOBS + col] = dnn[r];
    lik_sol[row * NOBS + col] = dsol[r];
  }
}

// ---------------------------------------------------------------------------
extern "C" void kernel_launch(void* const* d_in, const int* in_sizes, int n_in,
                              void* d_out, int out_size, void* d_ws,
                              size_t ws_size, hipStream_t stream) {
  const float* obs_loc = (const float*)d_in[0];
  const float* obs_val = (const float*)d_in[1];
  const float* theta0  = (const float*)d_in[2];
  const float* eps_o   = (const float*)d_in[3];
  const float* u_o     = (const float*)d_in[4];
  const float* eps_da  = (const float*)d_in[5];
  const float* u_da    = (const float*)d_in[6];

  float* out         = (float*)d_out;
  float* acc_list    = out;                       // [10000]
  float* theta_inner = out + ITER_DA;             // [10000][2]
  float* lik_nn      = out + ITER_DA * 3;         // [10000][256]
  float* lik_sol     = lik_nn + ITER_DA * NOBS;   // [10000][256]
  float* ws          = (float*)d_ws;              // 784 floats used

  // 1) basis + Gram sufficient statistics
  k_basis_gram<<<1, 256, 0, stream>>>(obs_loc, obs_val, ws);
  // 2) zero acc_list + theta_inner (unfilled rows must be 0; also the GEMM's
  //    zero-padded A matrix)
  k_zero<<<(ITER_DA * 3 + 255) / 256, 256, 0, stream>>>(out, ITER_DA * 3);
  // 3) serial chain (latency-bound; minimal per-step critical path)
  k_chain<<<1, 32, 0, stream>>>(ws, theta0, eps_o, u_o, eps_da, u_da,
                                acc_list, theta_inner);
  // 4) dense WMMA GEMM for both prediction matrices:
  //    625 M-tiles x 16 N-tiles = 10000 waves, 8 waves/block
  k_wmma_preds<<<1250, 256, 0, stream>>>(theta_inner, ws + 16, lik_nn,
                                         lik_sol);
}
